// MultiHeadAttention_2482491097407
// MI455X (gfx1250) — matrix operand order repaired and checked by reading, not compiled, not run on hardware
//
#include <hip/hip_runtime.h>

// ---------------- problem constants ----------------
constexpr int NH  = 16;     // heads
constexpr int NC  = 1024;   // hidden
constexpr int ND  = 64;     // head dim
constexpr int NB  = 2;      // batch
constexpr int NT  = 2048;   // seq len
constexpr int NBT = NB * NT;
constexpr float LN_EPS = 1e-5f;

// ---------------- types ----------------
typedef __attribute__((ext_vector_type(16))) __bf16   v16bf;
typedef __attribute__((ext_vector_type(8)))  float    v8f;
typedef __attribute__((ext_vector_type(4)))  unsigned u32x4;
typedef __attribute__((ext_vector_type(4)))  float    f32x4;

// ---------------- gfx1250 async-copy path (guarded) ----------------
#if defined(__has_builtin)
#if __has_builtin(__builtin_amdgcn_global_load_async_to_lds_b128)
#define USE_ASYNC_COPY 1
#endif
#endif

#ifdef USE_ASYNC_COPY
// builtin signature (from hipcc diagnostic): first param is
// addrspace(1) int4* (printed as "int __vector_size__(16) __device__ *"),
// i.e. (global src, lds dst, imm offset, imm cpol)
typedef int b128_t __attribute__((ext_vector_type(4)));
typedef __attribute__((address_space(1))) b128_t* as1_b128;
typedef __attribute__((address_space(3))) b128_t* as3_b128;
#define ASYNC_CP16(gsrc, ldst, off)                                   \
  __builtin_amdgcn_global_load_async_to_lds_b128(                     \
      (as1_b128)(gsrc), (as3_b128)(ldst), (off), 0)
#if __has_builtin(__builtin_amdgcn_s_wait_asynccnt)
#define WAIT_ASYNC() __builtin_amdgcn_s_wait_asynccnt(0)
#else
#define WAIT_ASYNC() asm volatile("s_wait_asynccnt 0x0" ::: "memory")
#endif
#endif

__device__ __forceinline__ unsigned short f2bf(float f) {
  unsigned u = __float_as_uint(f);
  u += 0x7FFFu + ((u >> 16) & 1u);          // round-to-nearest-even
  return (unsigned short)(u >> 16);
}
__device__ __forceinline__ unsigned pk2(float a, float b) {
  return (unsigned)f2bf(a) | ((unsigned)f2bf(b) << 16);
}

// A-operand fragment: per-lane two 16B chunks at +0 elems and +16 elems
__device__ __forceinline__ v16bf fragA(const unsigned short* p) {
  union { u32x4 u[2]; v16bf v; } x;
  x.u[0] = *(const u32x4*)(p);
  x.u[1] = *(const u32x4*)(p + 16);
  return x.v;
}
// B-operand fragment: per-lane column gather, rows +0..7 and +16..23 (stride ld)
__device__ __forceinline__ v16bf fragB(const unsigned short* p, int ld) {
  union { unsigned short s[16]; v16bf v; } x;
#pragma unroll
  for (int s = 0; s < 16; ++s) x.s[s] = p[(s < 8 ? s : s + 8) * ld];
  return x.v;
}
__device__ __forceinline__ v8f wmma_bf16(v16bf a, v16bf b, v8f c) {
  return __builtin_amdgcn_wmma_f32_16x16x32_bf16(false, a, false, b, (short)0, c,
                                                 false, false);
}
__device__ __forceinline__ v8f zero8() {
  v8f z;
#pragma unroll
  for (int i = 0; i < 8; ++i) z[i] = 0.0f;
  return z;
}

// =====================================================================
// Kernel 1: Yp = relu(X @ W + b).  z=0 -> Q (f32 + bf16), z=1 -> K^T
// per-head bf16, z=2 -> V row-major bf16.  128x128 tile / 256 threads.
// =====================================================================
__global__ __launch_bounds__(256)
void proj_kernel(const float* __restrict__ Qin, const float* __restrict__ Kin,
                 const float* __restrict__ Vin,
                 const float* __restrict__ Wq, const float* __restrict__ Wk,
                 const float* __restrict__ Wv,
                 const float* __restrict__ bq, const float* __restrict__ bk,
                 const float* __restrict__ bvp,
                 float* __restrict__ qp, unsigned short* __restrict__ qbf,
                 unsigned short* __restrict__ ktbf, unsigned short* __restrict__ vbf) {
  const int z = blockIdx.z;
  const float* X    = (z == 0) ? Qin : (z == 1) ? Kin : Vin;
  const float* W    = (z == 0) ? Wq  : (z == 1) ? Wk  : Wv;
  const float* bias = (z == 0) ? bq  : (z == 1) ? bk  : bvp;

  const int col0 = blockIdx.x * 128;
  const int row0 = blockIdx.y * 128;

  __shared__ alignas(16) unsigned short ldsA[128 * 40];  // (m,k) stride 40
  __shared__ alignas(16) unsigned short ldsB[32 * 136];  // (k,n) stride 136

  const int tid  = threadIdx.x;
  const int wid  = tid >> 5;
  const int lane = tid & 31;
  const int g    = (lane >> 4) & 1;
  const int l16  = lane & 15;
  const int wm   = wid & 3;   // wave row 0..3 (M)
  const int wn   = wid >> 2;  // wave col 0..1 (N)

  v8f acc[2][4];
#pragma unroll
  for (int i = 0; i < 2; ++i)
#pragma unroll
    for (int j = 0; j < 4; ++j) acc[i][j] = zero8();

  const int arow  = tid >> 1;         // 0..127
  const int akoff = (tid & 1) * 16;   // 0/16
  const int bkr   = tid >> 3;         // 0..31
  const int bnoff = (tid & 7) * 16;   // 0..112

  for (int kk = 0; kk < NC; kk += 32) {
    __syncthreads();
    {
      // ---- stage A tile: 128 x 32, f32 -> bf16 ----
      const float* pa = X + (size_t)(row0 + arow) * NC + kk + akoff;
      f32x4 f0 = *(const f32x4*)(pa + 0);
      f32x4 f1 = *(const f32x4*)(pa + 4);
      f32x4 f2 = *(const f32x4*)(pa + 8);
      f32x4 f3 = *(const f32x4*)(pa + 12);
      u32x4 u0, u1;
      u0.x = pk2(f0.x, f0.y); u0.y = pk2(f0.z, f0.w);
      u0.z = pk2(f1.x, f1.y); u0.w = pk2(f1.z, f1.w);
      u1.x = pk2(f2.x, f2.y); u1.y = pk2(f2.z, f2.w);
      u1.z = pk2(f3.x, f3.y); u1.w = pk2(f3.z, f3.w);
      *(u32x4*)&ldsA[arow * 40 + akoff]     = u0;
      *(u32x4*)&ldsA[arow * 40 + akoff + 8] = u1;

      // ---- stage B tile: 32 x 128, f32 -> bf16 (row-major, no transpose) ----
      const float* pb = W + (size_t)(kk + bkr) * NC + col0 + bnoff;
      f32x4 h0 = *(const f32x4*)(pb + 0);
      f32x4 h1 = *(const f32x4*)(pb + 4);
      f32x4 h2 = *(const f32x4*)(pb + 8);
      f32x4 h3 = *(const f32x4*)(pb + 12);
      u32x4 w0, w1;
      w0.x = pk2(h0.x, h0.y); w0.y = pk2(h0.z, h0.w);
      w0.z = pk2(h1.x, h1.y); w0.w = pk2(h1.z, h1.w);
      w1.x = pk2(h2.x, h2.y); w1.y = pk2(h2.z, h2.w);
      w1.z = pk2(h3.x, h3.y); w1.w = pk2(h3.z, h3.w);
      *(u32x4*)&ldsB[bkr * 136 + bnoff]     = w0;
      *(u32x4*)&ldsB[bkr * 136 + bnoff + 8] = w1;
    }
    __syncthreads();

    v16bf af[2], bf_[4];
#pragma unroll
    for (int i = 0; i < 2; ++i) {
      const int m = wm * 32 + i * 16 + l16;
      af[i] = fragA(&ldsA[m * 40 + g * 8]);
    }
    const int kr = l16 + 16 * g;  // lane's K-row
#pragma unroll
    for (int j = 0; j < 4; ++j)
      bf_[j] = fragB(&ldsB[(8 * g) * 136 + wn * 64 + j * 16 + l16], 136);

#pragma unroll
    for (int i = 0; i < 2; ++i)
#pragma unroll
      for (int j = 0; j < 4; ++j) acc[i][j] = wmma_bf16(af[i], bf_[j], acc[i][j]);
  }

  // ---- epilogue: bias + relu + scatter stores ----
#pragma unroll
  for (int i = 0; i < 2; ++i) {
#pragma unroll
    for (int j = 0; j < 4; ++j) {
      const int col = col0 + wn * 64 + j * 16 + l16;
      const float bcol = bias[col];
#pragma unroll
      for (int r = 0; r < 8; ++r) {
        const int row = row0 + wm * 32 + i * 16 + r + 8 * g;
        float v = acc[i][j][r] + bcol;
        v = v > 0.0f ? v : 0.0f;
        const size_t idx = (size_t)row * NC + col;
        if (z == 0) {
          qp[idx]  = v;
          qbf[idx] = f2bf(v);
        } else if (z == 2) {
          vbf[idx] = f2bf(v);
        } else {  // K -> transposed per head: kt[((b*NH+h)*ND+d)*NT + t]
          const int bb = row >> 11;          // / NT
          const int tq = row & (NT - 1);
          const int h  = col >> 6;
          const int d  = col & 63;
          ktbf[(((size_t)bb * NH + h) * ND + d) * NT + tq] = f2bf(v);
        }
      }
    }
  }
}

// =====================================================================
// Kernel 2: flash attention, one (b,h,64-query block) per 128-thread
// block; each of the 4 waves owns a 16-query tile.  Adds Qp residual.
// K/V tiles are staged with GLOBAL_LOAD_ASYNC_TO_LDS (ASYNCcnt) when
// the toolchain exposes the builtin; otherwise b128 VGPR copies.
// =====================================================================
__global__ __launch_bounds__(128)
void attn_kernel(const unsigned short* __restrict__ qbf,
                 const unsigned short* __restrict__ ktbf,
                 const unsigned short* __restrict__ vbf,
                 const float* __restrict__ qp, float* __restrict__ attn) {
  const int tid  = threadIdx.x;
  const int wid  = tid >> 5;
  const int lane = tid & 31;
  const int g    = (lane >> 4) & 1;
  const int l16  = lane & 15;

  const int h     = blockIdx.y;
  const int bb    = blockIdx.z;
  const int qbase = blockIdx.x * 64 + wid * 16;

  __shared__ alignas(16) unsigned short sKt[64 * 40];      // (d, key) stride 40
  __shared__ alignas(16) unsigned short sV[32 * 72];       // (key, d) stride 72
  __shared__ alignas(16) unsigned short sP[4][16 * 40];    // per-wave (q, key)

  // persistent Q fragments straight from global (contiguous 16B per lane)
  const unsigned short* qrow =
      qbf + ((size_t)(bb * NT + qbase + l16)) * NC + h * ND;
  const v16bf aq0 = fragA(qrow + g * 8);        // d in [0,32)
  const v16bf aq1 = fragA(qrow + 32 + g * 8);   // d in [32,64)

  v8f O[4];
#pragma unroll
  for (int nt = 0; nt < 4; ++nt) O[nt] = zero8();
  float m_i[8], l_i[8];
#pragma unroll
  for (int r = 0; r < 8; ++r) { m_i[r] = -INFINITY; l_i[r] = 0.0f; }

  const int sd  = tid >> 1;        // 0..63  (Kt row = d)
  const int sko = (tid & 1) * 16;  // key offset 0/16
  const int vk  = tid >> 2;        // 0..31  (V row = key)
  const int vdo = (tid & 3) * 16;  // d offset 0..48

  const unsigned short* ktb = ktbf + (((size_t)bb * NH + h) * ND) * NT;

  for (int k0 = 0; k0 < NT; k0 += 32) {
    __syncthreads();
    {
      const unsigned short* pk = ktb + (size_t)sd * NT + k0 + sko;
      unsigned short*       lk = &sKt[sd * 40 + sko];
      const unsigned short* pv =
          vbf + ((size_t)(bb * NT + k0 + vk)) * NC + h * ND + vdo;
      unsigned short*       lv = &sV[vk * 72 + vdo];
#ifdef USE_ASYNC_COPY
      ASYNC_CP16(pk, lk, 0);
      ASYNC_CP16(pk, lk, 16);
      ASYNC_CP16(pv, lv, 0);
      ASYNC_CP16(pv, lv, 16);
      WAIT_ASYNC();
#else
      u32x4 a0 = *(const u32x4*)pk;
      u32x4 a1 = *(const u32x4*)(pk + 8);
      *(u32x4*)(lk)     = a0;
      *(u32x4*)(lk + 8) = a1;
      u32x4 b0 = *(const u32x4*)pv;
      u32x4 b1 = *(const u32x4*)(pv + 8);
      *(u32x4*)(lv)     = b0;
      *(u32x4*)(lv + 8) = b1;
#endif
    }
    __syncthreads();

    // ---- scores: S = Q K^T for 32 keys (two 16-key tiles) ----
    v8f s0 = zero8(), s1 = zero8();
    {
      const int d0 = l16 + 16 * g;  // lane's contraction row for step 0
      v16bf b00 = fragB(&sKt[(8 * g)      * 40 + 0 + l16], 40);
      v16bf b01 = fragB(&sKt[(8 * g + 32) * 40 + 0 + l16], 40);
      v16bf b10 = fragB(&sKt[(8 * g)      * 40 + 16 + l16], 40);
      v16bf b11 = fragB(&sKt[(8 * g + 32) * 40 + 16 + l16], 40);
      s0 = wmma_bf16(aq0, b00, s0);
      s0 = wmma_bf16(aq1, b01, s0);
      s1 = wmma_bf16(aq0, b10, s1);
      s1 = wmma_bf16(aq1, b11, s1);
    }
#pragma unroll
    for (int e = 0; e < 8; ++e) { s0[e] *= 0.125f; s1[e] *= 0.125f; }

    // ---- online softmax (per-row stats via 16-lane shuffle reduce) ----
#pragma unroll
    for (int r = 0; r < 8; ++r) {
      float mx = fmaxf(s0[r], s1[r]);
      mx = fmaxf(mx, __shfl_xor(mx, 1));
      mx = fmaxf(mx, __shfl_xor(mx, 2));
      mx = fmaxf(mx, __shfl_xor(mx, 4));
      mx = fmaxf(mx, __shfl_xor(mx, 8));
      const float mn    = fmaxf(m_i[r], mx);
      const float alpha = __expf(m_i[r] - mn);
      m_i[r] = mn;
      const float p0 = __expf(s0[r] - mn);
      const float p1 = __expf(s1[r] - mn);
      float sum = p0 + p1;
      sum += __shfl_xor(sum, 1);
      sum += __shfl_xor(sum, 2);
      sum += __shfl_xor(sum, 4);
      sum += __shfl_xor(sum, 8);
      l_i[r] = l_i[r] * alpha + sum;
      O[0][r] *= alpha; O[1][r] *= alpha; O[2][r] *= alpha; O[3][r] *= alpha;
      const int prow = r + 8 * g;
      sP[wid][prow * 40 + l16]      = f2bf(p0);
      sP[wid][prow * 40 + 16 + l16] = f2bf(p1);
    }

    // ---- O += P @ V ----
    const v16bf ap = fragA(&sP[wid][l16 * 40 + g * 8]);
    const int kr = l16 + 16 * g;
#pragma unroll
    for (int nt = 0; nt < 4; ++nt) {
      v16bf bvf = fragB(&sV[(8 * g) * 72 + nt * 16 + l16], 72);
      O[nt] = wmma_bf16(ap, bvf, O[nt]);
    }
  }

  // ---- finalize: 1/l scale + residual (Qp) ----
#pragma unroll
  for (int r = 0; r < 8; ++r) {
    const float inv = 1.0f / l_i[r];
    const int q = qbase + r + 8 * g;
    const size_t off = ((size_t)(bb * NT + q)) * NC + h * ND;
#pragma unroll
    for (int nt = 0; nt < 4; ++nt) {
      const int d = nt * 16 + l16;
      attn[off + d] = O[nt][r] * inv + qp[off + d];
    }
  }
}

// =====================================================================
// Kernel 3: LayerNorm over C=1024 per row (256 threads, one f32x4 each)
// =====================================================================
__global__ __launch_bounds__(256)
void ln_kernel(const float* __restrict__ x, const float* __restrict__ gam,
               const float* __restrict__ bet, float* __restrict__ out) {
  const int row = blockIdx.x;
  const int tid = threadIdx.x;
  const float* xr = x + (size_t)row * NC;
  const f32x4 f = *(const f32x4*)(xr + tid * 4);
  float s  = f.x + f.y + f.z + f.w;
  float ss = f.x * f.x + f.y * f.y + f.z * f.z + f.w * f.w;
#pragma unroll
  for (int m = 1; m < 32; m <<= 1) {
    s  += __shfl_xor(s, m);
    ss += __shfl_xor(ss, m);
  }
  __shared__ float sh[16];
  const int wv = tid >> 5, ln = tid & 31;
  if (ln == 0) { sh[wv] = s; sh[8 + wv] = ss; }
  __syncthreads();
  float ts = 0.0f, tss = 0.0f;
#pragma unroll
  for (int i = 0; i < 8; ++i) { ts += sh[i]; tss += sh[8 + i]; }
  const float mean = ts * (1.0f / NC);
  const float var  = tss * (1.0f / NC) - mean * mean;
  const float rstd = rsqrtf(var + LN_EPS);
  const int c0 = tid * 4;
  f32x4 o;
  o.x = gam[c0 + 0] * (f.x - mean) * rstd + bet[c0 + 0];
  o.y = gam[c0 + 1] * (f.y - mean) * rstd + bet[c0 + 1];
  o.z = gam[c0 + 2] * (f.z - mean) * rstd + bet[c0 + 2];
  o.w = gam[c0 + 3] * (f.w - mean) * rstd + bet[c0 + 3];
  *(f32x4*)(out + (size_t)row * NC + c0) = o;
}

// =====================================================================
extern "C" void kernel_launch(void* const* d_in, const int* in_sizes, int n_in,
                              void* d_out, int out_size, void* d_ws,
                              size_t ws_size, hipStream_t stream) {
  (void)in_sizes; (void)n_in; (void)out_size; (void)ws_size;
  const float* Qin = (const float*)d_in[0];
  const float* Kin = (const float*)d_in[1];
  const float* Vin = (const float*)d_in[2];
  // d_in[3], d_in[4]: pad masks -- mathematical no-ops in the reference
  const float* Wq  = (const float*)d_in[5];
  const float* bq  = (const float*)d_in[6];
  const float* Wk  = (const float*)d_in[7];
  const float* bk  = (const float*)d_in[8];
  const float* Wv  = (const float*)d_in[9];
  const float* bv  = (const float*)d_in[10];
  const float* lng = (const float*)d_in[11];
  const float* lnb = (const float*)d_in[12];

  const size_t NE = (size_t)NBT * NC;  // 4M elements
  char* ws = (char*)d_ws;
  float*          qp   = (float*)(ws);                        // 16 MB
  unsigned short* qbf  = (unsigned short*)(ws + NE * 4);      //  8 MB
  unsigned short* ktbf = (unsigned short*)(ws + NE * 6);      //  8 MB
  unsigned short* vbf  = (unsigned short*)(ws + NE * 8);      //  8 MB
  float*          atn  = (float*)(ws + NE * 10);              // 16 MB

  dim3 gp(NC / 128, NBT / 128, 3);
  proj_kernel<<<gp, 256, 0, stream>>>(Qin, Kin, Vin, Wq, Wk, Wv, bq, bk, bv,
                                      qp, qbf, ktbf, vbf);
  dim3 ga(NT / 64, NH, NB);
  attn_kernel<<<ga, 128, 0, stream>>>(qbf, ktbf, vbf, qp, atn);
  ln_kernel<<<NBT, 256, 0, stream>>>(atn, lng, lnb, (float*)d_out);
}
